// DecoderRNN_49727131353142
// MI455X (gfx1250) — matrix operand order repaired and checked by reading, not compiled, not run on hardware
//
#include <hip/hip_runtime.h>
#include <hip/hip_bf16.h>

typedef __attribute__((ext_vector_type(16))) _Float16 v16h;
typedef __attribute__((ext_vector_type(8)))  _Float16 v8h;
typedef __attribute__((ext_vector_type(8)))  float    v8f;

#define B_    32
#define T_    20
#define P_    196      // HH*WW
#define DENC_ 256
#define E_    512
#define H_    512
#define A_    256
#define V_    30000
#define XK_   1280     // E + DENC + H (combined LSTM input)
#define G4H_  2048     // 4*H

union AFrag { v16h v; v8h h[2]; };

__device__ __forceinline__ float sigm(float x) { return 1.0f / (1.0f + __expf(-x)); }

// ---------------------------------------------------------------------------
// Generic WMMA GEMM: D[M,N] = A[M,K] (f16, row-major) * Bt[N,K]^T (f16) + bias[n]
// One wave computes one 16x16 tile; f32 accumulation via v_wmma_f32_16x16x32_f16.
// Fragment layouts per CDNA5 ISA 7.12.2:
//   A  : lane L holds row m=L%16; elems 0..7 -> k=8*(L/16)+0..7, elems 8..15 -> +16
//   B  : lane L holds col n=L%16; elems 0..7 -> k=8*(L/16)+0..7, elems 8..15 -> +16
//   C/D: lane L, vgpr r -> m=r+8*(L/16), n=L%16
// ---------------------------------------------------------------------------
__global__ void wmma_gemm(const _Float16* __restrict__ A,
                          const _Float16* __restrict__ Bt,
                          float* __restrict__ D,
                          const float* __restrict__ bias,
                          int M, int N, int K, int ldd) {
  const int wavesPerBlock = blockDim.x >> 5;
  int tile = blockIdx.x * wavesPerBlock + (threadIdx.x >> 5);
  int nTiles = N >> 4;
  int mTiles = M >> 4;
  if (tile >= mTiles * nTiles) return;
  int tm = tile / nTiles;
  int tn = tile - tm * nTiles;
  int lane = threadIdx.x & 31;
  int half = lane >> 4;
  int l16  = lane & 15;
  int mrow = (tm << 4) + l16;
  int nrow = (tn << 4) + l16;

  float bv = bias ? bias[nrow] : 0.0f;
  v8f c;
#pragma unroll
  for (int r = 0; r < 8; ++r) c[r] = bv;

  const _Float16* arow = A  + (size_t)mrow * K + (half << 3);
  const _Float16* brow = Bt + (size_t)nrow * K + (half << 3);

  for (int k0 = 0; k0 < K; k0 += 32) {
    AFrag a, b;
    a.h[0] = *(const v8h*)(arow + k0);
    a.h[1] = *(const v8h*)(arow + k0 + 16);
    b.h[0] = *(const v8h*)(brow + k0);
    b.h[1] = *(const v8h*)(brow + k0 + 16);
    c = __builtin_amdgcn_wmma_f32_16x16x32_f16(false, a.v, false, b.v,
                                               (short)0, c, false, false);
  }

  int mbase = (tm << 4) + (half << 3);
  float* drow = D + (size_t)mbase * ldd + nrow;
#pragma unroll
  for (int r = 0; r < 8; ++r) drow[(size_t)r * ldd] = c[r];
}

// ---------------------------------------------------------------------------
// Preprocessing kernels
// ---------------------------------------------------------------------------

// dst[n*K + k] = (f16) src[k*N + n]   (transpose + cast of [K,N] weight)
__global__ void convert_transpose(const float* __restrict__ src,
                                  _Float16* __restrict__ dst, int K, int N) {
  long long idx = (long long)blockIdx.x * blockDim.x + threadIdx.x;
  if (idx >= (long long)N * K) return;
  int n = (int)(idx / K);
  int k = (int)(idx - (long long)n * K);
  dst[idx] = (_Float16)src[(size_t)k * N + n];
}

// wcomb[n*XK + k]: k<768 -> W_ih[k][n], else W_hh[k-768][n]   (transposed f16)
__global__ void build_wcomb(const float* __restrict__ W_ih,
                            const float* __restrict__ W_hh,
                            _Float16* __restrict__ dst) {
  long long idx = (long long)blockIdx.x * blockDim.x + threadIdx.x;
  if (idx >= (long long)G4H_ * XK_) return;
  int n = (int)(idx / XK_);
  int k = (int)(idx - (long long)n * XK_);
  float v = (k < E_ + DENC_) ? W_ih[(size_t)k * G4H_ + n]
                             : W_hh[(size_t)(k - (E_ + DENC_)) * G4H_ + n];
  dst[idx] = (_Float16)v;
}

__global__ void build_bcomb(const float* __restrict__ b_ih,
                            const float* __restrict__ b_hh,
                            float* __restrict__ dst) {
  int i = blockIdx.x * blockDim.x + threadIdx.x;
  if (i < G4H_) dst[i] = b_ih[i] + b_hh[i];
}

__global__ void cast_f16(const float* __restrict__ src, _Float16* __restrict__ dst,
                         long long n) {
  long long idx = (long long)blockIdx.x * blockDim.x + threadIdx.x;
  if (idx < n) dst[idx] = (_Float16)src[idx];
}

// xf16[(t*B + b)*E + e] = (f16) emb[captions[b*T + t]][e]
__global__ void gather_emb(const float* __restrict__ emb,
                           const int* __restrict__ captions,
                           _Float16* __restrict__ xf16) {
  long long idx = (long long)blockIdx.x * blockDim.x + threadIdx.x;
  if (idx >= (long long)T_ * B_ * E_) return;
  int t = (int)(idx / (B_ * E_));
  int r = (int)(idx - (long long)t * (B_ * E_));
  int b = r / E_;
  int e = r - b * E_;
  int cap = captions[b * T_ + t];
  xf16[idx] = (_Float16)emb[(size_t)cap * E_ + e];
}

// ---------------------------------------------------------------------------
// Per-step kernels
// ---------------------------------------------------------------------------

// copy x_t (pre-gathered f16 embedding) into xcat cols [0, E)
__global__ void set_x(const _Float16* __restrict__ xf16,
                      _Float16* __restrict__ xcat, int t) {
  int idx = blockIdx.x * blockDim.x + threadIdx.x;
  if (idx >= B_ * E_) return;
  int b = idx / E_;
  int e = idx - b * E_;
  xcat[(size_t)b * XK_ + e] = xf16[((size_t)t * B_ + b) * E_ + e];
}

// Fused attention: e = tanh(att1 + att2)·W_full + b_full ; softmax over P ;
// context = alpha·enc ; write context (f16) into xcat cols [E, E+DENC)
__global__ void escore_softmax_ctx(const float* __restrict__ att1,
                                   const float* __restrict__ att2,
                                   const float* __restrict__ wfull,
                                   const float* __restrict__ bfull,
                                   const float* __restrict__ enc,
                                   _Float16* __restrict__ xcat) {
  int b = blockIdx.x;
  int tid = threadIdx.x;               // 256 threads
  __shared__ float a2[A_];
  __shared__ float red[256];
  __shared__ float al[256];

  a2[tid] = att2[(size_t)b * A_ + tid];
  __syncthreads();

  bool act = tid < P_;
  float ev = 0.0f;
  if (act) {
    const float* row = att1 + ((size_t)b * P_ + tid) * A_;
    float acc = bfull[0];
    for (int a = 0; a < A_; ++a)
      acc += tanhf(row[a] + a2[a]) * wfull[a];
    ev = acc;
  }

  red[tid] = act ? ev : -3.402823466e38f;
  __syncthreads();
  for (int s = 128; s > 0; s >>= 1) {
    if (tid < s) red[tid] = fmaxf(red[tid], red[tid + s]);
    __syncthreads();
  }
  float mx = red[0];
  __syncthreads();

  float ex = act ? __expf(ev - mx) : 0.0f;
  red[tid] = ex;
  __syncthreads();
  for (int s = 128; s > 0; s >>= 1) {
    if (tid < s) red[tid] += red[tid + s];
    __syncthreads();
  }
  float inv = 1.0f / red[0];
  al[tid] = ex * inv;
  __syncthreads();

  // context dim d = tid (DENC_ == 256 == blockDim)
  float ctx = 0.0f;
  const float* eb = enc + (size_t)b * P_ * DENC_ + tid;
  for (int p = 0; p < P_; ++p) ctx += al[p] * eb[(size_t)p * DENC_];
  xcat[(size_t)b * XK_ + E_ + tid] = (_Float16)ctx;
}

// LSTM pointwise: update c, produce h (f16) into hbuf and xcat cols [E+DENC, XK)
__global__ void lstm_pointwise(const float* __restrict__ gates,
                               float* __restrict__ cbuf,
                               _Float16* __restrict__ hbuf,
                               _Float16* __restrict__ xcat) {
  int idx = blockIdx.x * blockDim.x + threadIdx.x;
  if (idx >= B_ * H_) return;
  int b = idx / H_;
  int n = idx - b * H_;
  const float* g = gates + (size_t)b * G4H_;
  float gi = g[n];
  float gf = g[H_ + n];
  float gg = g[2 * H_ + n];
  float go = g[3 * H_ + n];
  float c  = cbuf[idx];
  float cn = sigm(gf) * c + sigm(gi) * tanhf(gg);
  float hn = sigm(go) * tanhf(cn);
  cbuf[idx] = cn;
  _Float16 hh = (_Float16)hn;
  hbuf[idx] = hh;
  xcat[(size_t)b * XK_ + (E_ + DENC_) + n] = hh;
}

// ---------------------------------------------------------------------------
// Host-side launcher
// ---------------------------------------------------------------------------
extern "C" void kernel_launch(void* const* d_in, const int* in_sizes, int n_in,
                              void* d_out, int out_size, void* d_ws, size_t ws_size,
                              hipStream_t stream) {
  const float* encoder_out = (const float*)d_in[0];
  const int*   captions    = (const int*)  d_in[1];
  const float* W_enc_att   = (const float*)d_in[2];
  const float* b_enc_att   = (const float*)d_in[3];
  const float* W_dec_att   = (const float*)d_in[4];
  const float* b_dec_att   = (const float*)d_in[5];
  const float* W_full      = (const float*)d_in[6];
  const float* b_full      = (const float*)d_in[7];
  const float* emb         = (const float*)d_in[8];
  const float* W_ih        = (const float*)d_in[9];
  const float* b_ih        = (const float*)d_in[10];
  const float* W_hh        = (const float*)d_in[11];
  const float* b_hh        = (const float*)d_in[12];
  const float* W_fc        = (const float*)d_in[13];
  const float* b_fc        = (const float*)d_in[14];
  float* out = (float*)d_out;

  char* ws = (char*)d_ws;
  size_t off = 0;
  auto alloc = [&](size_t bytes) -> void* {
    void* p = ws + off;
    off += (bytes + 255) & ~(size_t)255;
    return p;
  };

  _Float16* wfc_t   = (_Float16*)alloc((size_t)V_ * H_ * sizeof(_Float16));
  _Float16* wcomb_t = (_Float16*)alloc((size_t)G4H_ * XK_ * sizeof(_Float16));
  _Float16* wdec_t  = (_Float16*)alloc((size_t)A_ * H_ * sizeof(_Float16));
  _Float16* wenc_t  = (_Float16*)alloc((size_t)A_ * DENC_ * sizeof(_Float16));
  float*    bcomb   = (float*)   alloc((size_t)G4H_ * sizeof(float));
  _Float16* enc_h   = (_Float16*)alloc((size_t)B_ * P_ * DENC_ * sizeof(_Float16));
  float*    att1    = (float*)   alloc((size_t)B_ * P_ * A_ * sizeof(float));
  _Float16* xf16    = (_Float16*)alloc((size_t)T_ * B_ * E_ * sizeof(_Float16));
  float*    att2    = (float*)   alloc((size_t)B_ * A_ * sizeof(float));
  _Float16* xcat    = (_Float16*)alloc((size_t)B_ * XK_ * sizeof(_Float16));
  _Float16* hbuf    = (_Float16*)alloc((size_t)B_ * H_ * sizeof(_Float16));
  float*    cbuf    = (float*)   alloc((size_t)B_ * H_ * sizeof(float));
  float*    gates   = (float*)   alloc((size_t)B_ * G4H_ * sizeof(float));

  const int thr = 256;
  auto blocks = [](long long n, int t) { return (int)((n + t - 1) / t); };
  auto gemmBlocks = [](long long tiles) { return (int)((tiles + 7) / 8); };

  // ---- preprocessing (state init + weight conversion) ----
  hipMemsetAsync(hbuf, 0, (size_t)B_ * H_ * sizeof(_Float16), stream);
  hipMemsetAsync(cbuf, 0, (size_t)B_ * H_ * sizeof(float), stream);
  hipMemsetAsync(xcat, 0, (size_t)B_ * XK_ * sizeof(_Float16), stream);

  convert_transpose<<<blocks((long long)A_ * DENC_, thr), thr, 0, stream>>>(
      W_enc_att, wenc_t, DENC_, A_);
  convert_transpose<<<blocks((long long)A_ * H_, thr), thr, 0, stream>>>(
      W_dec_att, wdec_t, H_, A_);
  convert_transpose<<<blocks((long long)V_ * H_, thr), thr, 0, stream>>>(
      W_fc, wfc_t, H_, V_);
  build_wcomb<<<blocks((long long)G4H_ * XK_, thr), thr, 0, stream>>>(
      W_ih, W_hh, wcomb_t);
  build_bcomb<<<blocks(G4H_, thr), thr, 0, stream>>>(b_ih, b_hh, bcomb);
  cast_f16<<<blocks((long long)B_ * P_ * DENC_, thr), thr, 0, stream>>>(
      encoder_out, enc_h, (long long)B_ * P_ * DENC_);
  gather_emb<<<blocks((long long)T_ * B_ * E_, thr), thr, 0, stream>>>(
      emb, captions, xf16);

  // att1 = enc @ W_enc_att + b_enc_att   [6272,256] x [256,256]
  {
    long long tiles = (long long)((B_ * P_) / 16) * (A_ / 16);
    wmma_gemm<<<gemmBlocks(tiles), thr, 0, stream>>>(
        enc_h, wenc_t, att1, b_enc_att, B_ * P_, A_, DENC_, A_);
  }

  // ---- sequential timestep loop ----
  for (int t = 0; t < T_; ++t) {
    set_x<<<blocks(B_ * E_, thr), thr, 0, stream>>>(xf16, xcat, t);

    // att2 = h @ W_dec_att + b_dec_att   [32,512] x [512,256]
    wmma_gemm<<<gemmBlocks((B_ / 16) * (A_ / 16)), thr, 0, stream>>>(
        hbuf, wdec_t, att2, b_dec_att, B_, A_, H_, A_);

    escore_softmax_ctx<<<B_, 256, 0, stream>>>(att1, att2, W_full, b_full,
                                               encoder_out, xcat);

    // gates = [x, ctx, h] @ [W_ih; W_hh] + (b_ih + b_hh)   [32,1280] x [1280,2048]
    wmma_gemm<<<gemmBlocks((B_ / 16) * (G4H_ / 16)), thr, 0, stream>>>(
        xcat, wcomb_t, gates, bcomb, B_, G4H_, XK_, G4H_);

    lstm_pointwise<<<blocks(B_ * H_, thr), thr, 0, stream>>>(gates, cbuf, hbuf,
                                                             xcat);

    // out[:, t, :] = h_new @ W_fc + b_fc   [32,512] x [512,30000]
    wmma_gemm<<<gemmBlocks((long long)(B_ / 16) * (V_ / 16)), thr, 0, stream>>>(
        hbuf, wfc_t, out + (size_t)t * V_, b_fc, B_, V_, H_, (int)(T_ * V_));
  }
}
